// LRULayer_37744172597608
// MI455X (gfx1250) — matrix operand order repaired and checked by reading, not compiled, not run on hardware
//
#include <hip/hip_runtime.h>

// ---------------- problem constants ----------------
#define HID   768          // hidden dim H
#define PSS   384          // SSM size P
#define LSEQ  4096         // sequence length
#define NBAT  8            // batch
#define MTOT  (NBAT*LSEQ)  // 32768 tokens
#define NCHUNK 16          // scan chunks
#define LCH   256          // chunk length (NCHUNK*LCH == LSEQ)
#define LDSW  40           // padded LDS row stride (bf16 elems): 64B row + 16B pad

typedef __attribute__((ext_vector_type(16))) __bf16 v16bf;
typedef __attribute__((ext_vector_type(8)))  float  v8f;
typedef unsigned int v4u __attribute__((ext_vector_type(4)));
typedef int          v8i __attribute__((ext_vector_type(8)));
typedef int          v4i __attribute__((ext_vector_type(4)));

union Frag { v16bf v; uint4 q[2]; };

__device__ __forceinline__ unsigned short f2bf(float f) {
  union { float f; unsigned int u; } x; x.f = f;
  unsigned int u = x.u;
  u += 0x7FFFu + ((u >> 16) & 1u);          // round-to-nearest-even
  return (unsigned short)(u >> 16);
}

__device__ __forceinline__ unsigned int lds_off(const void* p) {
  // low 32 bits of a generic pointer to LDS == wave-relative LDS byte address
  return (unsigned int)(size_t)p;
}

// ---- Tensor Data Mover: load a 128-row x 32-col bf16 tile (row stride 768
// elems) into LDS with padded rows (64B data + 16B pad == LDSW elems). ----
__device__ __forceinline__ void tdm_load_tile(const unsigned short* gsrc,
                                              unsigned int ldsaddr,
                                              unsigned int tensor_rows) {
  unsigned long long ga = (unsigned long long)(size_t)gsrc;
  v4u g0;
  g0[0] = 1u;                                        // count=1, user desc
  g0[1] = ldsaddr;                                   // lds_addr (bytes)
  g0[2] = (unsigned int)ga;                          // global_addr[31:0]
  g0[3] = (unsigned int)(ga >> 32) | (2u << 30);     // global_addr[56:32] | type=2
  v8i g1;
  // wg_mask=0 | data_size=1(2B) | pad_en | pad_interval=3(64B) | pad_amount=3(16B)
  g1[0] = (int)((1u << 16) | (1u << 20) | (3u << 22) | (3u << 25));
  g1[1] = (int)(768u << 16);                         // tensor_dim0 = 768 (lo16)
  g1[2] = (int)((tensor_rows & 0xFFFFu) << 16);      // dim0 hi=0 | tensor_dim1 lo16
  g1[3] = (int)((tensor_rows >> 16) | (32u << 16));  // tensor_dim1 hi16 | tile_dim0=32
  g1[4] = (int)128u;                                 // tile_dim1=128 | tile_dim2=0
  g1[5] = (int)768;                                  // tensor_dim0_stride = 768
  g1[6] = 0;
  g1[7] = 0;
  v4i z = {0, 0, 0, 0};
#if __clang_major__ >= 23
  v8i z8 = {0, 0, 0, 0, 0, 0, 0, 0};
  __builtin_amdgcn_tensor_load_to_lds(g0, g1, z, z, z8, 0);
#else
  __builtin_amdgcn_tensor_load_to_lds(g0, g1, z, z, 0);
#endif
}

// A fragment (16x32 bf16, MxK): lane r=l&15 -> row m=r; hi=l>>4 selects
// K in {hi*8..hi*8+7} U {16+hi*8..23+hi*8}  (ISA 7.12.2 16-bit A layout)
__device__ __forceinline__ void load_fragA(const unsigned short* lds, int row, int hi, Frag& f) {
  const unsigned short* p0 = lds + row * LDSW + hi * 8;
  f.q[0] = *(const uint4*)(p0);
  f.q[1] = *(const uint4*)(p0 + 16);
}

// B fragment (32x16 bf16, KxN): lane r=l&15 -> col n=r; hi=l>>4 selects
// K in {hi*8..hi*8+7} U {16+hi*8..23+hi*8}  (same K order as A)
__device__ __forceinline__ void load_fragB(const unsigned short* lds, int row, int hi, Frag& f) {
  const unsigned short* p0 = lds + row * LDSW + hi * 8;
  f.q[0] = *(const uint4*)(p0);
  f.q[1] = *(const uint4*)(p0 + 16);
}

// ---------------- parameter prep ----------------
__global__ void k_prep_lam(const float* __restrict__ nu_log,
                           const float* __restrict__ theta_log,
                           float* lam_re, float* lam_im,
                           float* lp_re,  float* lp_im) {
  int p = threadIdx.x;
  if (p >= PSS) return;
  float nu = __expf(nu_log[p]);
  float th = __expf(theta_log[p]);
  float r  = __expf(-nu);
  float lre = r * __cosf(th);
  float lim = r * __sinf(th);
  lam_re[p] = lre; lam_im[p] = lim;
  float ar = lre, ai = lim;
#pragma unroll
  for (int i = 0; i < 8; ++i) {             // lam^(2^8) = lam^256
    float nr = ar * ar - ai * ai;
    float ni = 2.f * ar * ai;
    ar = nr; ai = ni;
  }
  lp_re[p] = ar; lp_im[p] = ai;
}

// W1[n,h] = B[p,h,part]*gamma[p]  (n<384 -> re, else im);  W2[h,n] = +-C[h,p,part]
__global__ void k_prep_w(const float* __restrict__ Bp, const float* __restrict__ Cp,
                         const float* __restrict__ gamma_log,
                         unsigned short* __restrict__ W1, unsigned short* __restrict__ W2) {
  int idx = blockIdx.x * blockDim.x + threadIdx.x;
  if (idx >= HID * HID) return;
  int n = idx / HID, h = idx % HID;
  int part = (n >= PSS) ? 1 : 0;
  int p = n - part * PSS;
  float g = __expf(gamma_log[p]);
  W1[idx] = f2bf(Bp[(size_t)(p * HID + h) * 2 + part] * g);

  int h2 = idx / HID, n2 = idx % HID;
  int part2 = (n2 >= PSS) ? 1 : 0;
  int p2 = n2 - part2 * PSS;
  float v = Cp[(size_t)(h2 * PSS + p2) * 2 + part2];
  W2[idx] = f2bf(part2 ? -v : v);
}

// ---------------- GEMM1: bu[m,n] = fp32 inputs x bf16 W1 ----------------
// Double-buffered LDS: A staged by threads (fp32->bf16 convert), B via TDM.
__global__ void __launch_bounds__(256)
k_gemm1(const float* __restrict__ X, const unsigned short* __restrict__ W,
        float* __restrict__ bu) {
  __shared__ unsigned short As[2][128 * LDSW];
  __shared__ unsigned short Bs[2][128 * LDSW];
  const int tid  = threadIdx.x;
  const int lane = tid & 31, wave = tid >> 5;
  const int wm = (wave >> 1) * 32;      // 0,32,64,96
  const int wn = (wave & 1) * 64;       // 0,64
  const int r  = lane & 15, hi = lane >> 4;
  const int gm0 = blockIdx.x * 128;
  const int gn0 = blockIdx.y * 128;

  v8f acc[2][4];
#pragma unroll
  for (int i = 0; i < 2; ++i)
#pragma unroll
    for (int j = 0; j < 4; ++j)
#pragma unroll
      for (int e = 0; e < 8; ++e) acc[i][j][e] = 0.f;

  const int cc = tid & 7;               // float4 within row
  const int r0 = tid >> 3;              // 0..31
  auto stage_A = [&](int k0, int buf) {
#pragma unroll
    for (int rr = 0; rr < 4; ++rr) {
      const int row = r0 + rr * 32;
      const float4 v = *(const float4*)(X + (size_t)(gm0 + row) * HID + k0 + cc * 4);
      union { unsigned short s[4]; uint2 u; } t;
      t.s[0] = f2bf(v.x); t.s[1] = f2bf(v.y); t.s[2] = f2bf(v.z); t.s[3] = f2bf(v.w);
      *(uint2*)(&As[buf][row * LDSW + cc * 4]) = t.u;
    }
  };

  stage_A(0, 0);
  if (wave == 0)
    tdm_load_tile(W + (size_t)gn0 * HID, lds_off(&Bs[0][0]), HID);

  const int nK = HID / 32;              // 24
  for (int i = 0; i < nK; ++i) {
    if (wave == 0) __builtin_amdgcn_s_wait_tensorcnt(0);
    __syncthreads();                    // publishes buf i&1 (A ds-stores + B TDM)
    const int buf = i & 1;
    if (i + 1 < nK) {
      stage_A((i + 1) * 32, buf ^ 1);
      if (wave == 0)
        tdm_load_tile(W + (size_t)gn0 * HID + (i + 1) * 32,
                      lds_off(&Bs[buf ^ 1][0]), HID);
      __builtin_prefetch(X + (size_t)(gm0 + r0) * HID + (i + 2) * 32 + cc * 4, 0, 1);
    }
    Frag a[2], b[4];
#pragma unroll
    for (int mt = 0; mt < 2; ++mt) load_fragA(&As[buf][0], wm + mt * 16 + r, hi, a[mt]);
#pragma unroll
    for (int nt = 0; nt < 4; ++nt) load_fragB(&Bs[buf][0], wn + nt * 16 + r, hi, b[nt]);
#pragma unroll
    for (int mt = 0; mt < 2; ++mt)
#pragma unroll
      for (int nt = 0; nt < 4; ++nt)
        acc[mt][nt] = __builtin_amdgcn_wmma_f32_16x16x32_bf16(
            false, a[mt].v, false, b[nt].v, (short)0, acc[mt][nt], false, false);
  }

#pragma unroll
  for (int mt = 0; mt < 2; ++mt)
#pragma unroll
    for (int nt = 0; nt < 4; ++nt) {
      const int n = gn0 + wn + nt * 16 + r;
#pragma unroll
      for (int i = 0; i < 8; ++i) {
        const int m = gm0 + wm + mt * 16 + i + 8 * hi;
        bu[(size_t)m * HID + n] = acc[mt][nt][i];
      }
    }
}

// ---------------- scan stage 1: local 256-step scans ----------------
__global__ void k_scan1(float* __restrict__ bu,
                        const float* __restrict__ lam_re, const float* __restrict__ lam_im,
                        float* __restrict__ carry) {
  const int c = blockIdx.x, pt = blockIdx.y, b = blockIdx.z;
  const int p = pt * 128 + threadIdx.x;
  const float lre = lam_re[p], lim = lam_im[p];
  float xr = 0.f, xi = 0.f;
  size_t base = ((size_t)(b * LSEQ + c * LCH)) * HID + p;
  for (int j = 0; j < LCH; ++j) {
    float br = bu[base], bi = bu[base + PSS];
    float nr = fmaf(lre, xr, fmaf(-lim, xi, br));
    float ni = fmaf(lre, xi, fmaf(lim, xr, bi));
    xr = nr; xi = ni;
    bu[base] = xr; bu[base + PSS] = xi;
    base += HID;
  }
  int ci = (b * NCHUNK + c) * PSS + p;
  carry[ci] = xr;
  carry[ci + NBAT * NCHUNK * PSS] = xi;
}

// ---------------- scan stage 2: carry scan across 16 chunks ----------------
__global__ void k_scan2(const float* __restrict__ carry,
                        const float* __restrict__ lp_re, const float* __restrict__ lp_im,
                        float* __restrict__ prefix) {
  const int pt = blockIdx.x, b = blockIdx.z;
  const int p = pt * 128 + threadIdx.x;
  const float ar = lp_re[p], ai = lp_im[p];   // lam^256
  float xr = 0.f, xi = 0.f;
  for (int c = 0; c < NCHUNK; ++c) {
    int ci = (b * NCHUNK + c) * PSS + p;
    float cr = carry[ci], cim = carry[ci + NBAT * NCHUNK * PSS];
    float nr = ar * xr - ai * xi + cr;
    float ni = ar * xi + ai * xr + cim;
    xr = nr; xi = ni;
    prefix[ci] = xr;
    prefix[ci + NBAT * NCHUNK * PSS] = xi;
  }
}

// ---------------- scan stage 3: fixup + convert states to bf16 ----------------
__global__ void k_scan3(const float* __restrict__ bu,
                        const float* __restrict__ lam_re, const float* __restrict__ lam_im,
                        const float* __restrict__ prefix, unsigned short* __restrict__ S) {
  const int c = blockIdx.x, pt = blockIdx.y, b = blockIdx.z;
  const int p = pt * 128 + threadIdx.x;
  const float lre = lam_re[p], lim = lam_im[p];
  float pr = 0.f, pi = 0.f;
  if (c > 0) {
    int ci = (b * NCHUNK + c - 1) * PSS + p;
    pr = prefix[ci]; pi = prefix[ci + NBAT * NCHUNK * PSS];
  }
  float fr = lre, fi = lim;                    // lam^(j+1)
  size_t base = ((size_t)(b * LSEQ + c * LCH)) * HID + p;
  for (int j = 0; j < LCH; ++j) {
    float xr = bu[base]       + fr * pr - fi * pi;
    float xi = bu[base + PSS] + fr * pi + fi * pr;
    S[base]       = f2bf(xr);
    S[base + PSS] = f2bf(xi);
    float nfr = fr * lre - fi * lim;
    float nfi = fr * lim + fi * lre;
    fr = nfr; fi = nfi;
    base += HID;
  }
}

// ---------------- GEMM2: out = bf16 states x bf16 W2 + inputs*D ----------------
// Double-buffered LDS; both A (states) and B (weights) tiles via TDM.
__global__ void __launch_bounds__(256)
k_gemm2(const unsigned short* __restrict__ S, const unsigned short* __restrict__ W,
        const float* __restrict__ X, const float* __restrict__ Dv,
        float* __restrict__ out) {
  __shared__ unsigned short As[2][128 * LDSW];
  __shared__ unsigned short Bs[2][128 * LDSW];
  const int tid  = threadIdx.x;
  const int lane = tid & 31, wave = tid >> 5;
  const int wm = (wave >> 1) * 32;
  const int wn = (wave & 1) * 64;
  const int r  = lane & 15, hi = lane >> 4;
  const int gm0 = blockIdx.x * 128;
  const int gn0 = blockIdx.y * 128;

  v8f acc[2][4];
#pragma unroll
  for (int i = 0; i < 2; ++i)
#pragma unroll
    for (int j = 0; j < 4; ++j)
#pragma unroll
      for (int e = 0; e < 8; ++e) acc[i][j][e] = 0.f;

  if (wave == 0) {
    tdm_load_tile(S + (size_t)gm0 * HID, lds_off(&As[0][0]), MTOT);
    tdm_load_tile(W + (size_t)gn0 * HID, lds_off(&Bs[0][0]), HID);
  }

  const int nK = HID / 32;              // 24
  for (int i = 0; i < nK; ++i) {
    if (wave == 0) __builtin_amdgcn_s_wait_tensorcnt(0);
    __syncthreads();                    // publishes buf i&1
    const int buf = i & 1;
    if (i + 1 < nK && wave == 0) {
      const int k1 = (i + 1) * 32;
      tdm_load_tile(S + (size_t)gm0 * HID + k1, lds_off(&As[buf ^ 1][0]), MTOT);
      tdm_load_tile(W + (size_t)gn0 * HID + k1, lds_off(&Bs[buf ^ 1][0]), HID);
    }
    Frag a[2], b[4];
#pragma unroll
    for (int mt = 0; mt < 2; ++mt) load_fragA(&As[buf][0], wm + mt * 16 + r, hi, a[mt]);
#pragma unroll
    for (int nt = 0; nt < 4; ++nt) load_fragB(&Bs[buf][0], wn + nt * 16 + r, hi, b[nt]);
#pragma unroll
    for (int mt = 0; mt < 2; ++mt)
#pragma unroll
      for (int nt = 0; nt < 4; ++nt)
        acc[mt][nt] = __builtin_amdgcn_wmma_f32_16x16x32_bf16(
            false, a[mt].v, false, b[nt].v, (short)0, acc[mt][nt], false, false);
  }

#pragma unroll
  for (int mt = 0; mt < 2; ++mt)
#pragma unroll
    for (int nt = 0; nt < 4; ++nt) {
      const int n = gn0 + wn + nt * 16 + r;
      const float dc = Dv[n];
#pragma unroll
      for (int i = 0; i < 8; ++i) {
        const int m = gm0 + wm + mt * 16 + i + 8 * hi;
        const size_t o = (size_t)m * HID + n;
        out[o] = acc[mt][nt][i] + X[o] * dc;
      }
    }
}

// ---------------- host launcher ----------------
extern "C" void kernel_launch(void* const* d_in, const int* in_sizes, int n_in,
                              void* d_out, int out_size, void* d_ws, size_t ws_size,
                              hipStream_t stream) {
  (void)in_sizes; (void)n_in; (void)out_size; (void)ws_size;
  const float* X         = (const float*)d_in[0];  // [8,4096,768]
  const float* nu_log    = (const float*)d_in[1];  // [384]
  const float* theta_log = (const float*)d_in[2];  // [384]
  const float* gamma_log = (const float*)d_in[3];  // [384]
  const float* Bp        = (const float*)d_in[4];  // [384,768,2]
  const float* Cp        = (const float*)d_in[5];  // [768,384,2]
  const float* Dv        = (const float*)d_in[6];  // [768]
  float* out = (float*)d_out;

  char* w = (char*)d_ws;
  size_t o = 0;
  float*          bu     = (float*)(w + o);          o += (size_t)MTOT * HID * 4;   // 100.7 MB
  unsigned short* S      = (unsigned short*)(w + o); o += (size_t)MTOT * HID * 2;   //  50.3 MB
  unsigned short* W1     = (unsigned short*)(w + o); o += (size_t)HID * HID * 2;
  unsigned short* W2     = (unsigned short*)(w + o); o += (size_t)HID * HID * 2;
  float*          lam_re = (float*)(w + o);          o += PSS * 4;
  float*          lam_im = (float*)(w + o);          o += PSS * 4;
  float*          lp_re  = (float*)(w + o);          o += PSS * 4;
  float*          lp_im  = (float*)(w + o);          o += PSS * 4;
  float*          carry  = (float*)(w + o);          o += (size_t)NBAT * NCHUNK * PSS * 2 * 4;
  float*          prefix = (float*)(w + o);          o += (size_t)NBAT * NCHUNK * PSS * 2 * 4;

  k_prep_lam<<<1, 384, 0, stream>>>(nu_log, theta_log, lam_re, lam_im, lp_re, lp_im);
  k_prep_w<<<(HID * HID + 255) / 256, 256, 0, stream>>>(Bp, Cp, gamma_log, W1, W2);
  k_gemm1<<<dim3(MTOT / 128, HID / 128), 256, 0, stream>>>(X, W1, bu);
  k_scan1<<<dim3(NCHUNK, PSS / 128, NBAT), 128, 0, stream>>>(bu, lam_re, lam_im, carry);
  k_scan2<<<dim3(PSS / 128, 1, NBAT), 128, 0, stream>>>(carry, lp_re, lp_im, prefix);
  k_scan3<<<dim3(NCHUNK, PSS / 128, NBAT), 128, 0, stream>>>(bu, lam_re, lam_im, prefix, S);
  k_gemm2<<<dim3(MTOT / 128, HID / 128), 256, 0, stream>>>(S, W2, X, Dv, out);
}
